// HopfieldTSP_59407987638704
// MI455X (gfx1250) — matrix operand order repaired and checked by reading, not compiled, not run on hardware
//
#include <hip/hip_runtime.h>
#include <hip/hip_bf16.h>

#define NCITY 8192
#define NITER 1000

typedef __bf16 bf16_t;
typedef bf16_t v16bf __attribute__((ext_vector_type(16)));
typedef float  v8f   __attribute__((ext_vector_type(8)));

// ---------------------------------------------------------------------------
// W is stored TILE-SWIZZLED in bf16 (134 MB -> fits MI455X 192MB L2):
//   tile T = 16 consecutive rows; per K-step s (32 K values) a 1KB block at
//     elem offset (T<<17) + (s<<9)
//   inside the block, WMMA lane l = (row%16) + 16*g needs
//     elements 0-7  = W[row][s*32 +      g*8 .. +7]  stored at  l*8
//     elements 8-15 = W[row][s*32 + 16 + g*8 .. +7]  stored at  256 + l*8
//   so both A-fragment loads are fully coalesced 512B wave loads.
//
//   c[i] = 2*rowsum_fp32(adj[i]) - (float)bf16(-adj[i][i])  keeps the
//   dominant diagonal term in fp32:  y_i = bf16row_i . bf16x + c_i * x_i.
// ---------------------------------------------------------------------------
__device__ __forceinline__ size_t swz_index(int row, int jb /*mult of 8*/) {
    const int s = jb >> 5;
    const int c = jb & 31;           // 0,8,16,24
    const int m = c >> 4;            // which half of the 32-K step
    const int g = (c & 15) >> 3;     // half-wave group
    return (((size_t)(row >> 4)) << 17) + (((size_t)s) << 9)
         + (m << 8) + (((row & 15) + 16 * g) << 3);
}

__global__ void __launch_bounds__(256)
prep_w_kernel(const float* __restrict__ adj,
              bf16_t* __restrict__ W,
              float* __restrict__ c) {
    __shared__ float sred[256];
    __shared__ float sdiag;
    const int row = blockIdx.x;
    const int tid = threadIdx.x;
    const float* arow = adj + (size_t)row * NCITY;

    float sum = 0.0f;
    for (int jb = tid * 8; jb < NCITY; jb += 256 * 8) {
        union { uint4 u; bf16_t h[8]; } pk;
        #pragma unroll
        for (int e = 0; e < 8; ++e) {
            float a = arow[jb + e];
            sum += a;
            bf16_t wv = (bf16_t)(-a);
            pk.h[e] = wv;
            if (jb + e == row) sdiag = (float)wv;   // exactly one writer
        }
        *(uint4*)(W + swz_index(row, jb)) = pk.u;
    }
    sred[tid] = sum;
    __syncthreads();
    for (int s = 128; s > 0; s >>= 1) {
        if (tid < s) sred[tid] += sred[tid + s];
        __syncthreads();
    }
    if (tid == 0) c[row] = 2.0f * sred[0] - sdiag;
}

__global__ void __launch_bounds__(256)
prep_x_kernel(const float* __restrict__ x, bf16_t* __restrict__ x16) {
    int i = blockIdx.x * 256 + threadIdx.x;
    if (i < NCITY) x16[i] = (bf16_t)x[i];
}

// ---------------------------------------------------------------------------
// One Hopfield step: y = sign(W_bf16 @ x16 + c .* x32), written fp32 + bf16.
// Block = one 16-row tile, 8 waves x 1024-K slices, 32 WMMAs per wave.
// A fragments come from the swizzled W with perfectly coalesced b128 loads;
// B columns are 16 replicated copies of the x chunk (broadcast load), so all
// output columns are identical and lanes 0/16 carry the 16 row partials.
// The loop is pointer-increment form + unroll 8 so unrolled loads become
// immediate-offset b128s the compiler can clause and overlap with WMMAs.
// ---------------------------------------------------------------------------
__global__ void __launch_bounds__(256)
matvec_sign_kernel(const bf16_t* __restrict__ W,
                   const float*  __restrict__ c,
                   const float*  __restrict__ x32,
                   const bf16_t* __restrict__ x16,
                   float*        __restrict__ y32,
                   bf16_t*       __restrict__ y16) {
    __shared__ float red[8][16];

    const int tid  = threadIdx.x;
    const int wave = tid >> 5;
    const int lane = tid & 31;
    const int g    = lane >> 4;
    const int rowbase = blockIdx.x * 16;

    const int s0 = wave * 32;                  // 32 K-steps of 32 per wave
    const bf16_t* pa = W + (((size_t)blockIdx.x) << 17)
                         + (((size_t)s0) << 9) + lane * 8;
    const bf16_t* pb = x16 + (s0 << 5) + g * 8;

    v8f acc = {0.f, 0.f, 0.f, 0.f, 0.f, 0.f, 0.f, 0.f};

    #pragma unroll 8
    for (int it = 0; it < 32; ++it) {
        union { uint4 u[2]; v16bf v; } a;
        a.u[0] = *(const uint4*)(pa);          // coalesced 512B
        a.u[1] = *(const uint4*)(pa + 256);    // coalesced 512B
        union { uint4 u[2]; v16bf v; } bu;
        bu.u[0] = *(const uint4*)(pb);         // broadcast 2x16B/half-wave
        bu.u[1] = *(const uint4*)(pb + 16);
        v16bf b = bu.v;

        acc = __builtin_amdgcn_wmma_f32_16x16x32_bf16(
                  false, a.v, false, b, (short)0, acc, false, false);

        pa += 512;                             // next 1KB A block
        pb += 32;                              // next 32 x values
    }

    // Column 0 of C: lane 0 holds rows 0-7, lane 16 holds rows 8-15.
    if (lane == 0) {
        #pragma unroll
        for (int r = 0; r < 8; ++r) red[wave][r] = acc[r];
    } else if (lane == 16) {
        #pragma unroll
        for (int r = 0; r < 8; ++r) red[wave][8 + r] = acc[r];
    }
    __syncthreads();

    if (tid < 16) {
        const int r = rowbase + tid;
        float y = 0.0f;
        #pragma unroll
        for (int w = 0; w < 8; ++w) y += red[w][tid];
        y += c[r] * x32[r];                      // fp32 diagonal term
        float s = (y > 0.0f) ? 1.0f : ((y < 0.0f) ? -1.0f : 0.0f);
        y32[r] = s;
        y16[r] = (bf16_t)s;
    }
}

// ---------------------------------------------------------------------------
extern "C" void kernel_launch(void* const* d_in, const int* in_sizes, int n_in,
                              void* d_out, int out_size, void* d_ws, size_t ws_size,
                              hipStream_t stream) {
    const float* adj = (const float*)d_in[0];
    const float* x0  = (const float*)d_in[1];

    char* ws = (char*)d_ws;
    size_t off = 0;
    bf16_t* W  = (bf16_t*)(ws + off); off += (size_t)NCITY * NCITY * sizeof(bf16_t);
    float*  c  = (float*) (ws + off); off += (size_t)NCITY * sizeof(float);
    float*  xa = (float*) (ws + off); off += (size_t)NCITY * sizeof(float);
    float*  xb = (float*) (ws + off); off += (size_t)NCITY * sizeof(float);
    bf16_t* ha = (bf16_t*)(ws + off); off += (size_t)NCITY * sizeof(bf16_t);
    bf16_t* hb = (bf16_t*)(ws + off);

    prep_w_kernel<<<NCITY, 256, 0, stream>>>(adj, W, c);
    prep_x_kernel<<<NCITY / 256, 256, 0, stream>>>(x0, ha);

    float*  x32buf[2] = {xa, xb};
    bf16_t* x16buf[2] = {ha, hb};

    for (int t = 0; t < NITER; ++t) {
        const int rs  = t & 1;     // t=0 reads slot 0 (ha from prep, x32 from input)
        const int wsl = rs ^ 1;
        const float*  xin = (t == 0) ? x0 : x32buf[rs];
        const bf16_t* hin = x16buf[rs];
        float* xout = (t == NITER - 1) ? (float*)d_out : x32buf[wsl];
        matvec_sign_kernel<<<NCITY / 16, 256, 0, stream>>>(
            W, c, xin, hin, xout, x16buf[wsl]);
    }
}
